// Head_85761906967092
// MI455X (gfx1250) — matrix operand order repaired and checked by reading, not compiled, not run on hardware
//
#include <hip/hip_runtime.h>
#include <hip/hip_bf16.h>

#define B_  16
#define T_  2048
#define D_  1024
#define H_  64

typedef __attribute__((ext_vector_type(16))) __bf16 v16bf;
typedef __attribute__((ext_vector_type(8)))  __bf16 v8bf;
typedef __attribute__((ext_vector_type(8)))  float  v8f;

// ---------- helpers ----------

// Native fp32 -> bf16 conversion: lets clang use the hardware packed
// converter (v_cvt_pk_bf16_f32) instead of ~4 VALU ops of manual rounding.
static __device__ __forceinline__ __bf16 f2bf(float f) {
    return (__bf16)f;
}

static __device__ __forceinline__ v8f wmma_bf16(v16bf a, v16bf b, v8f c) {
    return __builtin_amdgcn_wmma_f32_16x16x32_bf16(
        /*neg_a=*/false, a, /*neg_b=*/false, b,
        /*c_mod=*/(short)0, c, /*reuse_a=*/false, /*reuse_b=*/false);
}

// A-fragment (16x32, MxK) from a row-major bf16 tile, leading dim ld (elements).
// ISA layout: row = lane&15 ; element i -> K = 16*(i>>3) + 8*(lane>>4) + (i&7)
static __device__ __forceinline__ v16bf load_afrag_bf(const __bf16* base, int ld) {
    int lane = threadIdx.x & 31;
    int row  = lane & 15;
    int c0   = (lane >> 4) * 8;
    const v8bf* lo = (const v8bf*)(base + (size_t)row * ld + c0);
    const v8bf* hi = (const v8bf*)(base + (size_t)row * ld + c0 + 16);
    v8bf a = *lo, b = *hi;
    v16bf r;
#pragma unroll
    for (int i = 0; i < 8; ++i) { r[i] = a[i]; r[8 + i] = b[i]; }
    return r;
}

// Same A-fragment but sourced from fp32 (converts to bf16).
static __device__ __forceinline__ v16bf load_afrag_f32(const float* base, int ld) {
    int lane = threadIdx.x & 31;
    int row  = lane & 15;
    int c0   = (lane >> 4) * 8;
    const float* p = base + (size_t)row * ld + c0;
    v16bf r;
#pragma unroll
    for (int i = 0; i < 8; ++i) { r[i] = f2bf(p[i]); r[8 + i] = f2bf(p[16 + i]); }
    return r;
}

// B-fragment (32x16, KxN) where memory is laid out N-major: mem[n*ld + k]
// ISA layout: col = lane&15 ; element i -> K = 16*(i>>3) + 8*(lane>>4) + (i&7)
static __device__ __forceinline__ v16bf load_bfrag(const __bf16* base, int ld) {
    int lane  = threadIdx.x & 31;
    int n     = lane & 15;
    int kbase = (lane >> 4) * 8;
    const v8bf* p = (const v8bf*)(base + (size_t)n * ld + kbase);
    v8bf a = p[0], b = p[2];
    v16bf r;
#pragma unroll
    for (int i = 0; i < 8; ++i) { r[i] = a[i]; r[8 + i] = b[i]; }
    return r;
}

// ---------- kernel 0: W[D,H] fp32 -> Wt[3][H][D] bf16 (transposed) ----------

__global__ void attn_prep_w(const float* __restrict__ Wq,
                            const float* __restrict__ Wk,
                            const float* __restrict__ Wv,
                            __bf16* __restrict__ Wt) {
    int idx = blockIdx.x * 256 + threadIdx.x;
    if (idx >= 3 * H_ * D_) return;
    int w   = idx / (H_ * D_);
    int rem = idx % (H_ * D_);
    int n   = rem / D_;
    int k   = rem % D_;
    const float* W = (w == 0) ? Wq : ((w == 1) ? Wk : Wv);
    Wt[idx] = f2bf(W[(size_t)k * H_ + n]);
}

// ---------- kernel 1: projection Q,K,V = x @ W (one wave per 16-row tile) ----------

__global__ __launch_bounds__(128) void attn_proj(
        const float* __restrict__ x, const __bf16* __restrict__ Wt,
        __bf16* __restrict__ Q, __bf16* __restrict__ K, __bf16* __restrict__ Vt) {
    int wave = (blockIdx.x * 128 + threadIdx.x) >> 5;   // 2048 waves
    int lane = threadIdx.x & 31;
    int b    = wave >> 7;          // T/16 = 128 tiles per batch
    int tt   = wave & 127;
    const float* xb = x + ((size_t)b * T_ + tt * 16) * D_;

    v8f aq[4] = {}, ak[4] = {}, av[4] = {};

    for (int ks = 0; ks < D_ / 32; ++ks) {
        if (ks + 1 < D_ / 32) {
            const float* pf = xb + (size_t)(lane & 15) * D_ + (ks + 1) * 32;
            __builtin_prefetch(pf, 0, 1);               // global_prefetch_b8
        }
        v16bf xa = load_afrag_f32(xb + ks * 32, D_);
#pragma unroll
        for (int nt = 0; nt < 4; ++nt) {
            v16bf bq = load_bfrag(Wt + ((size_t)(0 * H_ + nt * 16)) * D_ + ks * 32, D_);
            aq[nt] = wmma_bf16(xa, bq, aq[nt]);
            v16bf bk = load_bfrag(Wt + ((size_t)(1 * H_ + nt * 16)) * D_ + ks * 32, D_);
            ak[nt] = wmma_bf16(xa, bk, ak[nt]);
            v16bf bv = load_bfrag(Wt + ((size_t)(2 * H_ + nt * 16)) * D_ + ks * 32, D_);
            av[nt] = wmma_bf16(xa, bv, av[nt]);
        }
    }

    // D layout: row = j + 8*(lane>=16), col = lane&15
    int half8 = (lane >> 4) * 8;
    int n     = lane & 15;
#pragma unroll
    for (int nt = 0; nt < 4; ++nt) {
#pragma unroll
        for (int j = 0; j < 8; ++j) {
            int t = tt * 16 + j + half8;
            int h = nt * 16 + n;
            Q[((size_t)b * T_ + t) * H_ + h]  = f2bf(aq[nt][j]);
            K[((size_t)b * T_ + t) * H_ + h]  = f2bf(ak[nt][j]);
            Vt[((size_t)b * H_ + h) * T_ + t] = f2bf(av[nt][j]);   // transposed for P·V
        }
    }
}

// ---------- kernel 2: causal flash attention (one wave per 16 query rows) ----------

__global__ __launch_bounds__(128) void attn_flash(
        const __bf16* __restrict__ Q, const __bf16* __restrict__ K,
        const __bf16* __restrict__ Vt, float* __restrict__ out) {
    __shared__ __bf16 ptile[4][16 * 32];   // per-wave P staging (D-layout -> A-layout)

    int warp  = threadIdx.x >> 5;
    int lane  = threadIdx.x & 31;
    int wave  = blockIdx.x * 4 + warp;     // 2048 waves
    int b     = wave >> 7;
    int q0    = (wave & 127) * 16;
    int half8 = (lane >> 4) * 8;
    int n     = lane & 15;

    const __bf16* qb   = Q  + ((size_t)b * T_ + q0) * H_;
    const __bf16* kbb  = K  + (size_t)b * T_ * H_;
    const __bf16* vtb  = Vt + (size_t)b * H_ * T_;
    __bf16* tile = &ptile[warp][0];

    v16bf qa0 = load_afrag_bf(qb, H_);        // h = 0..31
    v16bf qa1 = load_afrag_bf(qb + 32, H_);   // h = 32..63

    v8f o0 = {}, o1 = {}, o2 = {}, o3 = {};
    float m_i[8], l_i[8];
#pragma unroll
    for (int j = 0; j < 8; ++j) { m_i[j] = -3.0e38f; l_i[j] = 0.0f; }
    const float scale = 0.125f;               // 1/sqrt(64)

    for (int key0 = 0; key0 <= q0 + 15; key0 += 32) {
        // S = Q · K^T for 16x32 key block (contraction over H=64 in two steps)
        v8f s0 = {}, s1 = {};
        s0 = wmma_bf16(qa0, load_bfrag(kbb + (size_t)key0 * H_,        H_), s0);
        s0 = wmma_bf16(qa1, load_bfrag(kbb + (size_t)key0 * H_ + 32,   H_), s0);
        s1 = wmma_bf16(qa0, load_bfrag(kbb + (size_t)(key0 + 16) * H_,      H_), s1);
        s1 = wmma_bf16(qa1, load_bfrag(kbb + (size_t)(key0 + 16) * H_ + 32, H_), s1);

        float alp[8];
#pragma unroll
        for (int j = 0; j < 8; ++j) {
            int qrow = q0 + j + half8;
            float a = s0[j] * scale; if (key0 + n      > qrow) a = -3.0e38f;
            float c = s1[j] * scale; if (key0 + 16 + n > qrow) c = -3.0e38f;
            // row max across the 16 lanes of this half-wave
            float mx = fmaxf(a, c);
            mx = fmaxf(mx, __shfl_xor(mx, 1, 32));
            mx = fmaxf(mx, __shfl_xor(mx, 2, 32));
            mx = fmaxf(mx, __shfl_xor(mx, 4, 32));
            mx = fmaxf(mx, __shfl_xor(mx, 8, 32));
            float mnew = fmaxf(m_i[j], mx);
            float al   = __expf(m_i[j] - mnew);
            m_i[j] = mnew;
            float p0 = __expf(a - mnew);
            float p1 = __expf(c - mnew);
            float rs = p0 + p1;
            rs += __shfl_xor(rs, 1, 32);
            rs += __shfl_xor(rs, 2, 32);
            rs += __shfl_xor(rs, 4, 32);
            rs += __shfl_xor(rs, 8, 32);
            l_i[j] = l_i[j] * al + rs;
            alp[j] = al;
            // stage P (D layout) into LDS as a plain row-major 16x32 bf16 tile
            tile[(j + half8) * 32 + n]      = f2bf(p0);
            tile[(j + half8) * 32 + 16 + n] = f2bf(p1);
        }
#pragma unroll
        for (int j = 0; j < 8; ++j) {
            o0[j] *= alp[j]; o1[j] *= alp[j]; o2[j] *= alp[j]; o3[j] *= alp[j];
        }
        // reload P in A-fragment striping (within-wave LDS RAW, DS ops in-order)
        v16bf pa = load_afrag_bf(tile, 32);
        // O += P · V  (V stored [h][t] so B-fragments are contiguous)
        o0 = wmma_bf16(pa, load_bfrag(vtb + (size_t)0  * T_ + key0, T_), o0);
        o1 = wmma_bf16(pa, load_bfrag(vtb + (size_t)16 * T_ + key0, T_), o1);
        o2 = wmma_bf16(pa, load_bfrag(vtb + (size_t)32 * T_ + key0, T_), o2);
        o3 = wmma_bf16(pa, load_bfrag(vtb + (size_t)48 * T_ + key0, T_), o3);
    }

    float* ob = out + ((size_t)b * T_ + q0) * H_;
#pragma unroll
    for (int j = 0; j < 8; ++j) {
        float inv = 1.0f / l_i[j];
        int t = j + half8;
        ob[(size_t)t * H_ + 0  + n] = o0[j] * inv;
        ob[(size_t)t * H_ + 16 + n] = o1[j] * inv;
        ob[(size_t)t * H_ + 32 + n] = o2[j] * inv;
        ob[(size_t)t * H_ + 48 + n] = o3[j] * inv;
    }
}

// ---------- launch ----------

extern "C" void kernel_launch(void* const* d_in, const int* in_sizes, int n_in,
                              void* d_out, int out_size, void* d_ws, size_t ws_size,
                              hipStream_t stream) {
    const float* x  = (const float*)d_in[0];
    const float* Wq = (const float*)d_in[1];
    const float* Wk = (const float*)d_in[2];
    const float* Wv = (const float*)d_in[3];

    const size_t WT_BYTES = (size_t)3 * H_ * D_ * 2;       // 384 KB
    const size_t QK_BYTES = (size_t)B_ * T_ * H_ * 2;      // 4 MB each
    const size_t OFF_WT = 0;
    const size_t OFF_Q  = OFF_WT + ((WT_BYTES + 255) & ~255ull);
    const size_t OFF_K  = OFF_Q + QK_BYTES;
    const size_t OFF_V  = OFF_K + QK_BYTES;
    if (ws_size < OFF_V + QK_BYTES) return;                // insufficient scratch

    char* ws = (char*)d_ws;
    __bf16* Wt = (__bf16*)(ws + OFF_WT);
    __bf16* Qb = (__bf16*)(ws + OFF_Q);
    __bf16* Kb = (__bf16*)(ws + OFF_K);
    __bf16* Vt = (__bf16*)(ws + OFF_V);

    attn_prep_w<<<(3 * H_ * D_ + 255) / 256, 256, 0, stream>>>(Wq, Wk, Wv, Wt);
    attn_proj  <<<(B_ * (T_ / 16)) / 4, 128, 0, stream>>>(x, Wt, Qb, Kb, Vt);
    attn_flash <<<(B_ * (T_ / 16)) / 4, 128, 0, stream>>>(Qb, Kb, Vt, (float*)d_out);
}
